// EnhancedGTATLayer_17686675325465
// MI455X (gfx1250) — compile-verified, matrix operand order checked by reading, not run on hardware
//
#include <hip/hip_runtime.h>
#include <hip/hip_bf16.h>
#include <stdint.h>

// ---------------------------------------------------------------------------
// EnhancedGTAT layer for MI455X (gfx1250), wave32 + WMMA bf16.
// All GEMMs run on v_wmma_f32_16x16x32_bf16 (f32 accumulate).
// B matrices pre-swizzled fragment-major: each lane's 16 B-values are one
// 32B chunk (two global_load_b128). A operands are bf16 so inner loops are
// pure b128-load + wmma. gcn/topo/res GEMMs fused (shared A), self-loop
// contribution folded into the GEMM epilogue.
// ---------------------------------------------------------------------------

#define NN   50000
#define EE   640000
#define CC   128          // CIN == COUT
#define HH   4
#define LNEPS 1e-5f

typedef __attribute__((ext_vector_type(16))) __bf16 bf16x16;
typedef __attribute__((ext_vector_type(8)))  float  floatx8;

// ---------------- WMMA fragment helpers (layouts per ISA 7.12.2) ----------

// A fragment from bf16 row-major: two contiguous 16B chunks per lane.
__device__ __forceinline__ bf16x16 load_a_frag_bf(const __bf16* __restrict__ A,
                                                  int lda, int row0, int k0) {
  const int lane = threadIdx.x & 31;
  const int m    = lane & 15;
  const int half = lane >> 4;
  const __bf16* p = A + (size_t)(row0 + m) * lda + k0 + half * 8;
  union { bf16x16 v; uint4 q[2]; } u;
  u.q[0] = *(const uint4*)(p);        // K = base+0..7
  u.q[1] = *(const uint4*)(p + 16);   // K = base+16..23
  return u.v;
}

// B fragment from pre-swizzled fragment-major storage:
// Bsw[((kt*NT + nt)*32 + lane)*16 + j]  -> one 32B contiguous chunk per lane.
__device__ __forceinline__ bf16x16 load_b_frag_sw(const __bf16* __restrict__ Bsw,
                                                  int NT, int kt, int nt) {
  const int lane = threadIdx.x & 31;
  const __bf16* p = Bsw + (((size_t)kt * NT + nt) * 32 + lane) * 16;
  union { bf16x16 v; uint4 q[2]; } u;
  u.q[0] = *(const uint4*)(p);
  u.q[1] = *(const uint4*)(p + 8);
  return u.v;
}

__device__ __forceinline__ floatx8 wmma_bf16(bf16x16 a, bf16x16 b, floatx8 c) {
  return __builtin_amdgcn_wmma_f32_16x16x32_bf16(false, a, false, b,
                                                 (short)0, c, false, false);
}

// fragment element j -> K offset within a 32-K tile, for lane "half"
__device__ __forceinline__ int frag_koff(int half, int j) {
  const int i = j >> 1, pos = j & 1;
  return (i < 4 ? half * 8 + 2 * i : 16 + half * 8 + 2 * (i - 4)) + pos;
}

// ---------------- weight packing kernels -----------------------------------

// row-major f32 [K x NC] -> fragment-major bf16 inside a dst with NTdst column
// tiles, placed starting at column-tile ntOff.
__global__ void k_pack_b(const float* __restrict__ src, __bf16* __restrict__ dst,
                         int K, int NC, int NTdst, int ntOff) {
  const int NT = NC >> 4;
  const int total = K * NC;
  for (int idx = blockIdx.x * blockDim.x + threadIdx.x; idx < total;
       idx += gridDim.x * blockDim.x) {
    const int j    = idx & 15;
    const int lane = (idx >> 4) & 31;
    const int tile = idx >> 9;          // kt*NT + nt (source tile order)
    const int nt = tile % NT;
    const int kt = tile / NT;
    const int half = lane >> 4, n = lane & 15;
    const int k = kt * 32 + frag_koff(half, j);
    dst[(((size_t)kt * NTdst + ntOff + nt) * 32 + lane) * 16 + j] =
        (__bf16)src[(size_t)k * NC + nt * 16 + n];
  }
}

// attn_W [H][2C][C]: logical B[c][h*128+d] (256 x 512) -> fragment-major bf16
__global__ void k_pack_attn(const float* __restrict__ src, __bf16* __restrict__ dst) {
  const int NC = HH * CC;               // 512
  const int NT = NC >> 4;               // 32
  const int total = 2 * CC * NC;        // 131072
  for (int idx = blockIdx.x * blockDim.x + threadIdx.x; idx < total;
       idx += gridDim.x * blockDim.x) {
    const int j    = idx & 15;
    const int lane = (idx >> 4) & 31;
    const int tile = idx >> 9;
    const int nt = tile % NT;
    const int kt = tile / NT;
    const int half = lane >> 4, n = lane & 15;
    const int c   = kt * 32 + frag_koff(half, j);
    const int col = nt * 16 + n;
    const int h = col >> 7, d = col & (CC - 1);
    dst[idx] = (__bf16)src[(size_t)h * (2 * CC * CC) + (size_t)c * CC + d];
  }
}

// ---------------- small utility kernels ------------------------------------

__global__ void k_f32_to_bf16(const float* __restrict__ s, __bf16* __restrict__ d,
                              int n) {
  for (int i = blockIdx.x * blockDim.x + threadIdx.x; i < n; i += gridDim.x * blockDim.x)
    d[i] = (__bf16)s[i];
}

__global__ void k_fill(float* __restrict__ p, float v, int n) {
  for (int i = blockIdx.x * blockDim.x + threadIdx.x; i < n; i += gridDim.x * blockDim.x)
    p[i] = v;
}

__global__ void k_deg_accum(const long long* __restrict__ ei, float* __restrict__ deg) {
  for (int e = blockIdx.x * blockDim.x + threadIdx.x; e < EE; e += gridDim.x * blockDim.x)
    atomicAdd(&deg[(int)ei[EE + e]], 1.0f);
}

__global__ void k_rsqrt(float* __restrict__ p, int n) {
  for (int i = blockIdx.x * blockDim.x + threadIdx.x; i < n; i += gridDim.x * blockDim.x)
    p[i] = rsqrtf(p[i]);
}

// one wave per edge; lane carries 4 consecutive floats (float4) -> 128/edge
__global__ void k_edge_scatter(const long long* __restrict__ ei,
                               const float* __restrict__ xw,
                               const float* __restrict__ dis,
                               float* __restrict__ agg) {
  const int wid    = (blockIdx.x * blockDim.x + threadIdx.x) >> 5;
  const int lane   = threadIdx.x & 31;
  const int nwaves = (gridDim.x * blockDim.x) >> 5;
  for (int e = wid; e < EE; e += nwaves) {
    const int s = (int)ei[e];
    const int d = (int)ei[EE + e];
    const float norm = dis[s] * dis[d];
    float4 v = ((const float4*)(xw + (size_t)s * CC))[lane];
    float* dst = agg + (size_t)d * CC + lane * 4;
    atomicAdd(dst + 0, v.x * norm);
    atomicAdd(dst + 1, v.y * norm);
    atomicAdd(dst + 2, v.z * norm);
    atomicAdd(dst + 3, v.w * norm);
  }
}

// row-wise LayerNorm(in + bias) -> relu -> f32 out (may alias in) + bf16 out
__global__ void k_ln_relu(const float* __restrict__ in, const float* __restrict__ bias,
                          const float* __restrict__ g, const float* __restrict__ b,
                          float* __restrict__ of, __bf16* __restrict__ ob) {
  const int wid    = (blockIdx.x * blockDim.x + threadIdx.x) >> 5;
  const int lane   = threadIdx.x & 31;
  const int nwaves = (gridDim.x * blockDim.x) >> 5;
  const float4 bb = ((const float4*)bias)[lane];
  const float4 gg = ((const float4*)g)[lane];
  const float4 be = ((const float4*)b)[lane];
  for (int row = wid; row < NN; row += nwaves) {
    float4 v = ((const float4*)(in + (size_t)row * CC))[lane];
    v.x += bb.x; v.y += bb.y; v.z += bb.z; v.w += bb.w;
    float s  = v.x + v.y + v.z + v.w;
    float s2 = v.x*v.x + v.y*v.y + v.z*v.z + v.w*v.w;
#pragma unroll
    for (int off = 16; off > 0; off >>= 1) {
      s  += __shfl_xor(s,  off, 32);
      s2 += __shfl_xor(s2, off, 32);
    }
    const float mean = s * (1.0f / CC);
    const float var  = s2 * (1.0f / CC) - mean * mean;
    const float rs   = rsqrtf(var + LNEPS);
    float4 o;
    o.x = fmaxf((v.x - mean) * rs * gg.x + be.x, 0.0f);
    o.y = fmaxf((v.y - mean) * rs * gg.y + be.y, 0.0f);
    o.z = fmaxf((v.z - mean) * rs * gg.z + be.z, 0.0f);
    o.w = fmaxf((v.w - mean) * rs * gg.w + be.w, 0.0f);
    ((float4*)(of + (size_t)row * CC))[lane] = o;
    __bf16* q = ob + (size_t)row * CC + lane * 4;
    q[0] = (__bf16)o.x; q[1] = (__bf16)o.y; q[2] = (__bf16)o.z; q[3] = (__bf16)o.w;
  }
}

// ---------------- WMMA GEMM kernels ----------------------------------------
// Block = 128 threads = 4 waves; each block owns 16 rows; N=50000 = 3125*16.

// Fused: [xw | topo_lin | res] = x_bf @ [gcnW | topoW | resW]  (128 x 384)
// Also writes agg = xw * dis[row]^2 (GCN self-loop term).
__global__ __launch_bounds__(128) void k_gemm3(const __bf16* __restrict__ xbf,
                                               const __bf16* __restrict__ Bsw,
                                               const float* __restrict__ res_b,
                                               const float* __restrict__ dis,
                                               float* __restrict__ xw,
                                               float* __restrict__ topo_f,
                                               float* __restrict__ resp,
                                               float* __restrict__ agg) {
  const int wave = threadIdx.x >> 5;
  const int row0 = blockIdx.x * 16;
  floatx8 acc[6];
#pragma unroll
  for (int t = 0; t < 6; ++t) acc[t] = (floatx8){};
#pragma unroll
  for (int kt = 0; kt < 4; ++kt) {
    bf16x16 a = load_a_frag_bf(xbf, CC, row0, kt * 32);
#pragma unroll
    for (int t = 0; t < 6; ++t) {
      bf16x16 b = load_b_frag_sw(Bsw, 24, kt, wave * 6 + t);
      acc[t] = wmma_bf16(a, b, acc[t]);
    }
  }
  const int lane = threadIdx.x & 31;
  const int n = lane & 15, half = lane >> 4;
#pragma unroll
  for (int r = 0; r < 8; ++r) {
    const int row = row0 + r + 8 * half;
    const float dd = dis[row];
    const float d2 = dd * dd;
#pragma unroll
    for (int t = 0; t < 6; ++t) {
      const int c = (wave * 6 + t) * 16 + n;   // 0..383 (tile-uniform branch)
      const float v = acc[t][r];
      if (c < CC) {
        xw [(size_t)row * CC + c] = v;
        agg[(size_t)row * CC + c] = v * d2;
      } else if (c < 2 * CC) {
        topo_f[(size_t)row * CC + (c - CC)] = v;
      } else {
        resp[(size_t)row * CC + (c - 2 * CC)] = v + res_b[c - 2 * CC];
      }
    }
  }
}

// gate = sigmoid([node||topo] @ attnW + attn_b); fused = g*node + (1-g)*topo
__global__ __launch_bounds__(128) void k_gate_fuse(const __bf16* __restrict__ node_bf,
                                                   const __bf16* __restrict__ topo_bf,
                                                   const float* __restrict__ node_f,
                                                   const float* __restrict__ topo_f,
                                                   const __bf16* __restrict__ attnWsw,
                                                   const float* __restrict__ attn_b,
                                                   __bf16* __restrict__ fused) {
  const int wave = threadIdx.x >> 5;
  const int row0 = blockIdx.x * 16;
  const int c0   = wave * 128;
  floatx8 acc[8];
#pragma unroll
  for (int t = 0; t < 8; ++t) acc[t] = (floatx8){};
  for (int kt = 0; kt < 8; ++kt) {
    const int kk = kt * 32;
    bf16x16 a = (kk < CC) ? load_a_frag_bf(node_bf, CC, row0, kk)
                          : load_a_frag_bf(topo_bf, CC, row0, kk - CC);
#pragma unroll
    for (int t = 0; t < 8; ++t) {
      bf16x16 b = load_b_frag_sw(attnWsw, 32, kt, wave * 8 + t);
      acc[t] = wmma_bf16(a, b, acc[t]);
    }
  }
  const int lane = threadIdx.x & 31;
  const int n = lane & 15, half = lane >> 4;
#pragma unroll
  for (int t = 0; t < 8; ++t) {
    const int cg = c0 + t * 16 + n;     // 0..511, == h*128 + d
    const int d  = cg & (CC - 1);
#pragma unroll
    for (int r = 0; r < 8; ++r) {
      const int row = row0 + r + 8 * half;
      const float gate = 1.0f / (1.0f + __expf(-(acc[t][r] + attn_b[cg])));
      const float nv = node_f[(size_t)row * CC + d];
      const float tv = topo_f[(size_t)row * CC + d];
      fused[(size_t)row * (HH * CC) + cg] = (__bf16)(gate * nv + (1.0f - gate) * tv);
    }
  }
}

// h1 = relu(fused_bf[N x 512] @ W1sw + b1) -> bf16
__global__ __launch_bounds__(128) void k_gemm_mlp1(const __bf16* __restrict__ A,
                                                   const __bf16* __restrict__ Bsw,
                                                   const float* __restrict__ bias,
                                                   __bf16* __restrict__ Hout) {
  const int wave = threadIdx.x >> 5;
  const int row0 = blockIdx.x * 16;
  const int c0   = wave * 32;
  const int nt0  = wave * 2;
  floatx8 acc0 = {}, acc1 = {};
#pragma unroll
  for (int kt = 0; kt < 16; ++kt) {
    bf16x16 a  = load_a_frag_bf(A, HH * CC, row0, kt * 32);
    bf16x16 b0 = load_b_frag_sw(Bsw, 8, kt, nt0);
    bf16x16 b1 = load_b_frag_sw(Bsw, 8, kt, nt0 + 1);
    acc0 = wmma_bf16(a, b0, acc0);
    acc1 = wmma_bf16(a, b1, acc1);
  }
  const int lane = threadIdx.x & 31;
  const int n = lane & 15, half = lane >> 4;
#pragma unroll
  for (int r = 0; r < 8; ++r) {
    const int row = row0 + r + 8 * half;
    Hout[(size_t)row * CC + c0 + n]      = (__bf16)fmaxf(acc0[r] + bias[c0 + n], 0.0f);
    Hout[(size_t)row * CC + c0 + 16 + n] = (__bf16)fmaxf(acc1[r] + bias[c0 + 16 + n], 0.0f);
  }
}

// out = relu(h1_bf @ W2sw + b2) + res_part   (res_part already has res_b)
__global__ __launch_bounds__(128) void k_gemm_mlp2_final(const __bf16* __restrict__ A,
                                                         const __bf16* __restrict__ Bsw,
                                                         const float* __restrict__ bias,
                                                         const float* __restrict__ resp,
                                                         float* __restrict__ Out) {
  const int wave = threadIdx.x >> 5;
  const int row0 = blockIdx.x * 16;
  const int c0   = wave * 32;
  const int nt0  = wave * 2;
  floatx8 acc0 = {}, acc1 = {};
#pragma unroll
  for (int kt = 0; kt < 4; ++kt) {
    bf16x16 a  = load_a_frag_bf(A, CC, row0, kt * 32);
    bf16x16 b0 = load_b_frag_sw(Bsw, 8, kt, nt0);
    bf16x16 b1 = load_b_frag_sw(Bsw, 8, kt, nt0 + 1);
    acc0 = wmma_bf16(a, b0, acc0);
    acc1 = wmma_bf16(a, b1, acc1);
  }
  const int lane = threadIdx.x & 31;
  const int n = lane & 15, half = lane >> 4;
#pragma unroll
  for (int r = 0; r < 8; ++r) {
    const int row = row0 + r + 8 * half;
    const size_t i0 = (size_t)row * CC + c0 + n;
    const size_t i1 = i0 + 16;
    Out[i0] = fmaxf(acc0[r] + bias[c0 + n], 0.0f)      + resp[i0];
    Out[i1] = fmaxf(acc1[r] + bias[c0 + 16 + n], 0.0f) + resp[i1];
  }
}

// ---------------------------------------------------------------------------

extern "C" void kernel_launch(void* const* d_in, const int* in_sizes, int n_in,
                              void* d_out, int out_size, void* d_ws, size_t ws_size,
                              hipStream_t stream) {
  (void)in_sizes; (void)n_in; (void)out_size; (void)ws_size;
  const float*     x       = (const float*)d_in[0];
  const long long* ei      = (const long long*)d_in[1];   // int64 edge_index [2][E]
  const float*     gcn_W   = (const float*)d_in[2];
  const float*     gcn_b   = (const float*)d_in[3];
  const float*     ln_ng   = (const float*)d_in[4];
  const float*     ln_nb   = (const float*)d_in[5];
  const float*     topo_W  = (const float*)d_in[6];
  const float*     topo_b  = (const float*)d_in[7];
  const float*     ln_tg   = (const float*)d_in[8];
  const float*     ln_tb   = (const float*)d_in[9];
  const float*     attn_W  = (const float*)d_in[10];
  const float*     attn_b  = (const float*)d_in[11];
  const float*     mlp_W1  = (const float*)d_in[12];
  const float*     mlp_b1  = (const float*)d_in[13];
  const float*     mlp_W2  = (const float*)d_in[14];
  const float*     mlp_b2  = (const float*)d_in[15];
  const float*     res_W   = (const float*)d_in[16];
  const float*     res_b   = (const float*)d_in[17];
  float* out = (float*)d_out;

  // --- workspace carve-out (~206 MB) ---
  char* base = (char*)d_ws;
  size_t off = 0;
  auto take = [&](size_t bytes) -> char* {
    char* p = base + off;
    off += (bytes + 255) & ~(size_t)255;
    return p;
  };
  float*   deg     = (float*)  take((size_t)NN * 4);             // becomes dis after rsqrt
  float*   xw      = (float*)  take((size_t)NN * CC * 4);
  float*   agg     = (float*)  take((size_t)NN * CC * 4);        // -> node_feat f32 (in place)
  float*   topo_f  = (float*)  take((size_t)NN * CC * 4);        // lin -> topo_feat (in place)
  float*   resp    = (float*)  take((size_t)NN * CC * 4);
  __bf16*  x_bf    = (__bf16*) take((size_t)NN * CC * 2);
  __bf16*  node_bf = (__bf16*) take((size_t)NN * CC * 2);
  __bf16*  topo_bf = (__bf16*) take((size_t)NN * CC * 2);
  __bf16*  fusedbf = (__bf16*) take((size_t)NN * HH * CC * 2);
  __bf16*  h1_bf   = (__bf16*) take((size_t)NN * CC * 2);
  __bf16*  w3b     = (__bf16*) take((size_t)CC * 3 * CC * 2);    // [gcn|topo|res]
  __bf16*  attnWb  = (__bf16*) take((size_t)2 * CC * HH * CC * 2);
  __bf16*  m1Wb    = (__bf16*) take((size_t)HH * CC * CC * 2);
  __bf16*  m2Wb    = (__bf16*) take((size_t)CC * CC * 2);

  const int rowBlocks = NN / 16;   // 3125, N divides evenly

  // 1) weight packing to fragment-major bf16
  k_pack_b<<<64,  256, 0, stream>>>(gcn_W,  w3b,  CC, CC, 24, 0);
  k_pack_b<<<64,  256, 0, stream>>>(topo_W, w3b,  CC, CC, 24, 8);
  k_pack_b<<<64,  256, 0, stream>>>(res_W,  w3b,  CC, CC, 24, 16);
  k_pack_b<<<256, 256, 0, stream>>>(mlp_W1, m1Wb, HH * CC, CC, 8, 0);
  k_pack_b<<<64,  256, 0, stream>>>(mlp_W2, m2Wb, CC, CC, 8, 0);
  k_pack_attn<<<512, 256, 0, stream>>>(attn_W, attnWb);

  // 2) x -> bf16 once (shared A operand of the fused triple GEMM)
  k_f32_to_bf16<<<4096, 256, 0, stream>>>(x, x_bf, NN * CC);

  // 3) degrees (self-loops give deg>=1), then dis = deg^-1/2 in place
  k_fill<<<196, 256, 0, stream>>>(deg, 1.0f, NN);
  k_deg_accum<<<2500, 256, 0, stream>>>(ei, deg);
  k_rsqrt<<<196, 256, 0, stream>>>(deg, NN);

  // 4) fused triple GEMM: xw / topo_lin / res (+bias), plus agg self-loop term
  k_gemm3<<<rowBlocks, 128, 0, stream>>>(x_bf, w3b, res_b, deg,
                                         xw, topo_f, resp, agg);

  // 5) GCN edge scatter-add into agg
  k_edge_scatter<<<4096, 256, 0, stream>>>(ei, xw, deg, agg);

  // 6) LayerNorm + ReLU (writes f32 in place + bf16 copies for WMMA inputs)
  k_ln_relu<<<1024, 256, 0, stream>>>(agg,    gcn_b,  ln_ng, ln_nb, agg,    node_bf);
  k_ln_relu<<<1024, 256, 0, stream>>>(topo_f, topo_b, ln_tg, ln_tb, topo_f, topo_bf);

  // 7) gated fusion (256x512 GEMM + sigmoid + blend), then MLP + residual
  k_gate_fuse<<<rowBlocks, 128, 0, stream>>>(node_bf, topo_bf, agg, topo_f,
                                             attnWb, attn_b, fusedbf);
  k_gemm_mlp1<<<rowBlocks, 128, 0, stream>>>(fusedbf, m1Wb, mlp_b1, h1_bf);
  k_gemm_mlp2_final<<<rowBlocks, 128, 0, stream>>>(h1_bf, m2Wb, mlp_b2, resp, out);
}
